// EdgeEmbedder_42528766165472
// MI455X (gfx1250) — matrix operand order repaired and checked by reading, not compiled, not run on hardware
//
#include <hip/hip_runtime.h>
#include <hip/hip_bf16.h>

#define LSEQ 512
#define C_S 256
#define FEAT_DIM 64
#define NUM_BINS 22
#define C_PAIR_PRE 20
#define E_TOT 256          // total edge feature dim
#define C_P 128            // hidden / output dim
#define MAX_DIST 20.0f
#define ELD 264            // LDS stride (ushorts) for edge tile, padded
#define HLD 136            // LDS stride (ushorts) for hidden tile, padded

typedef __attribute__((ext_vector_type(16))) __bf16 v16bf;
typedef __attribute__((ext_vector_type(8)))  float  v8f;
typedef __attribute__((ext_vector_type(8)))  unsigned short ushort8;
typedef __attribute__((ext_vector_type(4)))  float  float4v;
struct ushort8x2 { ushort8 lo, hi; };

__device__ __forceinline__ unsigned short f2bf(float f) {
  unsigned u = __builtin_bit_cast(unsigned, f);
  u += 0x7FFFu + ((u >> 16) & 1u);       // round-to-nearest-even
  return (unsigned short)(u >> 16);
}

// A fragment, 16x32 bf16 (ISA 7.12.2: lanes 0-15 rows M, VGPR0-3 K=half*8+0..7,
// VGPR4-7 K=16+half*8+0..7)
__device__ __forceinline__ v16bf load_frag_a(const unsigned short* base, int ldk,
                                             int kbase, int lane) {
  const int row  = lane & 15;
  const int half = lane >> 4;
  const unsigned short* p = base + row * ldk + kbase + half * 8;
  ushort8x2 t;
  t.lo = *(const ushort8*)(p);
  t.hi = *(const ushort8*)(p + 16);
  return __builtin_bit_cast(v16bf, t);
}

// B fragment, 32x16 bf16: lane n holds column N=n%16, K=half*8+0..7 then K=16+half*8+0..7.
// W stored row-major [N][K] so a lane's K-values are two contiguous runs of 8.
__device__ __forceinline__ v16bf load_frag_b(const unsigned short* W, int ldk,
                                             int kbase, int lane) {
  const int col   = lane & 15;
  const int khalf = (lane >> 4) * 8;
  const unsigned short* p = W + col * ldk + kbase + khalf;
  ushort8x2 t;
  t.lo = *(const ushort8*)(p);
  t.hi = *(const ushort8*)(p + 16);
  return __builtin_bit_cast(v16bf, t);
}

// ---------------- setup kernels ----------------

__global__ void convw_kernel(const float* __restrict__ W1, const float* __restrict__ W2,
                             unsigned short* __restrict__ W1b, unsigned short* __restrict__ W2b) {
  int id = blockIdx.x * blockDim.x + threadIdx.x;
  if (id < C_P * E_TOT) W1b[id] = f2bf(W1[id]);
  else if (id < C_P * E_TOT + C_P * C_P) {
    int k = id - C_P * E_TOT;
    W2b[k] = f2bf(W2[k]);
  }
}

// p_i[l][f] = dot(s[l,:], W_sp[f,:]) + b_sp[f], stored bf16 (512x64)
__global__ void pi_kernel(const float* __restrict__ s, const float* __restrict__ W_sp,
                          const float* __restrict__ b_sp, unsigned short* __restrict__ pib) {
  __shared__ float srow[C_S];
  const int l = blockIdx.x;
  const int f = threadIdx.x;          // 64 threads
  for (int c = f; c < C_S; c += 64) srow[c] = s[(size_t)l * C_S + c];
  __syncthreads();
  const float* w = W_sp + (size_t)f * C_S;
  float acc = b_sp[f];
  for (int c = 0; c < C_S; ++c) acc = fmaf(srow[c], w[c], acc);
  pib[(size_t)l * FEAT_DIM + f] = f2bf(acc);
}

// relpos table: row idx in [0,1024), rel = idx-512;
// relT[idx][g] = dot(pos_emb(rel), W_rp[g,:]) + b_rp[g], bf16
__global__ void relpos_kernel(const float* __restrict__ W_rp, const float* __restrict__ b_rp,
                              unsigned short* __restrict__ relT) {
  __shared__ float emb[FEAT_DIM];
  const int idx = blockIdx.x;
  const float rel = (float)(idx - 512);
  const int e = threadIdx.x;          // 64 threads
  const int k = e & 31;
  const float ang = rel * 3.14159265358979323846f /
                    __powf(2056.0f, (2.0f * (float)k) / (float)FEAT_DIM);
  emb[e] = (e < 32) ? __sinf(ang) : __cosf(ang);
  __syncthreads();
  const float* w = W_rp + (size_t)e * FEAT_DIM;
  float acc = b_rp[e];
  for (int c = 0; c < FEAT_DIM; ++c) acc = fmaf(emb[c], w[c], acc);
  relT[(size_t)idx * FEAT_DIM + e] = f2bf(acc);
}

// ---------------- main fused kernel ----------------

__global__ __launch_bounds__(256) void edge_main_kernel(
    const float* __restrict__ t, const float* __restrict__ sc_t,
    const float* __restrict__ pre,
    const float* __restrict__ b1, const float* __restrict__ b2,
    const unsigned short* __restrict__ W1b, const unsigned short* __restrict__ W2b,
    const unsigned short* __restrict__ pib, const unsigned short* __restrict__ relT,
    float* __restrict__ out) {
  const int i  = blockIdx.y;
  const int jb = blockIdx.x * 128;
  const int tid = threadIdx.x;

  __shared__ alignas(16) unsigned short edge[128 * ELD];   // 128 x 256 bf16 tile
  __shared__ alignas(16) unsigned short hbuf[128 * HLD];   // 128 x 128 bf16 hidden
  __shared__ alignas(16) unsigned short pii[FEAT_DIM];
  __shared__ float ti[3], sci[3];
  __shared__ float b1s[C_P], b2s[C_P];

  // prefetch this block's pair_repr_pre rows (global_prefetch_b8)
  __builtin_prefetch(pre + (((size_t)i * LSEQ) + jb + (tid >> 1)) * C_PAIR_PRE, 0, 1);

  if (tid < FEAT_DIM) pii[tid] = pib[(size_t)i * FEAT_DIM + tid];
  if (tid < 3) { ti[tid] = t[i * 3 + tid]; sci[tid] = sc_t[i * 3 + tid]; }
  if (tid < C_P) b1s[tid] = b1[tid]; else b2s[tid - C_P] = b2[tid - C_P];
  __syncthreads();

  // ---- build the 128x256 bf16 all_edge tile: 2 threads per row ----
  {
    const int r = tid >> 1;
    const int j = jb + r;
    unsigned short* erow = edge + (size_t)r * ELD;
    if ((tid & 1) == 0) {
      // cols [0,64): p_i[i]; cols [64,128): p_i[j]
      ushort8* dst = (ushort8*)erow;
      const ushort8* srcA = (const ushort8*)pii;
      const ushort8* srcB = (const ushort8*)(pib + (size_t)j * FEAT_DIM);
#pragma unroll
      for (int c = 0; c < 8; ++c) dst[c] = srcA[c];
#pragma unroll
      for (int c = 0; c < 8; ++c) dst[8 + c] = srcB[c];
    } else {
      // cols [128,192): relpos table row
      ushort8* dst = (ushort8*)(erow + 2 * FEAT_DIM);
      const ushort8* rr = (const ushort8*)(relT + (size_t)(i - j + 512) * FEAT_DIM);
#pragma unroll
      for (int c = 0; c < 8; ++c) dst[c] = rr[c];
      // cols [192,214): RBF(||t_i - t_j||); [214,236): RBF(||sc_i - sc_j||)
      const float inv_sigma = (float)NUM_BINS / MAX_DIST;
      {
        float dx = ti[0] - t[j * 3 + 0];
        float dy = ti[1] - t[j * 3 + 1];
        float dz = ti[2] - t[j * 3 + 2];
        float d = sqrtf(dx * dx + dy * dy + dz * dz);
#pragma unroll
        for (int b = 0; b < NUM_BINS; ++b) {
          float mu = MAX_DIST * (float)b * (1.0f / (float)(NUM_BINS - 1));
          float u = (d - mu) * inv_sigma;
          erow[192 + b] = f2bf(__expf(-u * u));
        }
      }
      {
        float dx = sci[0] - sc_t[j * 3 + 0];
        float dy = sci[1] - sc_t[j * 3 + 1];
        float dz = sci[2] - sc_t[j * 3 + 2];
        float d = sqrtf(dx * dx + dy * dy + dz * dz);
#pragma unroll
        for (int b = 0; b < NUM_BINS; ++b) {
          float mu = MAX_DIST * (float)b * (1.0f / (float)(NUM_BINS - 1));
          float u = (d - mu) * inv_sigma;
          erow[214 + b] = f2bf(__expf(-u * u));
        }
      }
      // cols [236,256): pair_repr_pre, vectorized 5x float4 (80B, 16B-aligned)
      const float4v* pr4 = (const float4v*)(pre + (((size_t)i * LSEQ) + j) * C_PAIR_PRE);
#pragma unroll
      for (int c4 = 0; c4 < 5; ++c4) {
        float4v v = pr4[c4];
        erow[236 + c4 * 4 + 0] = f2bf(v.x);
        erow[236 + c4 * 4 + 1] = f2bf(v.y);
        erow[236 + c4 * 4 + 2] = f2bf(v.z);
        erow[236 + c4 * 4 + 3] = f2bf(v.w);
      }
    }
  }
  __syncthreads();

  const int lane = tid & 31;
  const int wave = tid >> 5;                 // 8 waves, each owns 16 M-rows
  const int rsub = (lane >> 4) * 8;
  const int n0   = lane & 15;

  // ---- GEMM1: (16 x 256) x (256 x 128) per wave, bf16 WMMA, f32 acc ----
  // B fragments fetched 4-at-a-time so multiple loads are in flight per WMMA group.
  v8f acc[8] = {};
  const unsigned short* arow = edge + (size_t)wave * 16 * ELD;
#pragma unroll
  for (int k = 0; k < 8; ++k) {
    v16bf a = load_frag_a(arow, ELD, k * 32, lane);
#pragma unroll
    for (int ph = 0; ph < 2; ++ph) {
      const int p0 = ph * 4;
      v16bf bf0 = load_frag_b(W1b + (size_t)(p0 + 0) * 16 * E_TOT, E_TOT, k * 32, lane);
      v16bf bf1 = load_frag_b(W1b + (size_t)(p0 + 1) * 16 * E_TOT, E_TOT, k * 32, lane);
      v16bf bf2 = load_frag_b(W1b + (size_t)(p0 + 2) * 16 * E_TOT, E_TOT, k * 32, lane);
      v16bf bf3 = load_frag_b(W1b + (size_t)(p0 + 3) * 16 * E_TOT, E_TOT, k * 32, lane);
      acc[p0 + 0] = __builtin_amdgcn_wmma_f32_16x16x32_bf16(
          false, a, false, bf0, (short)0, acc[p0 + 0], false, false);
      acc[p0 + 1] = __builtin_amdgcn_wmma_f32_16x16x32_bf16(
          false, a, false, bf1, (short)0, acc[p0 + 1], false, false);
      acc[p0 + 2] = __builtin_amdgcn_wmma_f32_16x16x32_bf16(
          false, a, false, bf2, (short)0, acc[p0 + 2], false, false);
      acc[p0 + 3] = __builtin_amdgcn_wmma_f32_16x16x32_bf16(
          false, a, false, bf3, (short)0, acc[p0 + 3], false, false);
    }
  }
  // bias + ReLU -> bf16 hidden tile
#pragma unroll
  for (int p = 0; p < 8; ++p) {
    const int col = p * 16 + n0;
    const float bias = b1s[col];
#pragma unroll
    for (int v = 0; v < 8; ++v) {
      float x = acc[p][v] + bias;
      x = x > 0.0f ? x : 0.0f;
      hbuf[(size_t)(wave * 16 + rsub + v) * HLD + col] = f2bf(x);
    }
  }
  __syncthreads();

  // ---- GEMM2: (16 x 128) x (128 x 128) per wave ----
  v8f acc2[8] = {};
  const unsigned short* hrow = hbuf + (size_t)wave * 16 * HLD;
#pragma unroll
  for (int k = 0; k < 4; ++k) {
    v16bf a = load_frag_a(hrow, HLD, k * 32, lane);
#pragma unroll
    for (int qh = 0; qh < 2; ++qh) {
      const int q0 = qh * 4;
      v16bf bf0 = load_frag_b(W2b + (size_t)(q0 + 0) * 16 * C_P, C_P, k * 32, lane);
      v16bf bf1 = load_frag_b(W2b + (size_t)(q0 + 1) * 16 * C_P, C_P, k * 32, lane);
      v16bf bf2 = load_frag_b(W2b + (size_t)(q0 + 2) * 16 * C_P, C_P, k * 32, lane);
      v16bf bf3 = load_frag_b(W2b + (size_t)(q0 + 3) * 16 * C_P, C_P, k * 32, lane);
      acc2[q0 + 0] = __builtin_amdgcn_wmma_f32_16x16x32_bf16(
          false, a, false, bf0, (short)0, acc2[q0 + 0], false, false);
      acc2[q0 + 1] = __builtin_amdgcn_wmma_f32_16x16x32_bf16(
          false, a, false, bf1, (short)0, acc2[q0 + 1], false, false);
      acc2[q0 + 2] = __builtin_amdgcn_wmma_f32_16x16x32_bf16(
          false, a, false, bf2, (short)0, acc2[q0 + 2], false, false);
      acc2[q0 + 3] = __builtin_amdgcn_wmma_f32_16x16x32_bf16(
          false, a, false, bf3, (short)0, acc2[q0 + 3], false, false);
    }
  }
  // bias + f32 store (coalesced across lanes over channel dim)
#pragma unroll
  for (int q = 0; q < 8; ++q) {
    const int col = q * 16 + n0;
    const float bias = b2s[col];
#pragma unroll
    for (int v = 0; v < 8; ++v) {
      const int jj = jb + wave * 16 + rsub + v;
      out[(((size_t)i * LSEQ) + jj) * C_P + col] = acc2[q][v] + bias;
    }
  }
}

// ---------------- host launch ----------------

extern "C" void kernel_launch(void* const* d_in, const int* in_sizes, int n_in,
                              void* d_out, int out_size, void* d_ws, size_t ws_size,
                              hipStream_t stream) {
  (void)in_sizes; (void)n_in; (void)out_size; (void)ws_size;
  // setup_inputs order:
  const float* s    = (const float*)d_in[0];
  const float* t    = (const float*)d_in[1];
  const float* sc_t = (const float*)d_in[2];
  const float* pre  = (const float*)d_in[3];
  // d_in[4] = p_mask (unused by reference)
  const float* W_sp = (const float*)d_in[5];
  const float* b_sp = (const float*)d_in[6];
  const float* W_rp = (const float*)d_in[7];
  const float* b_rp = (const float*)d_in[8];
  const float* W1   = (const float*)d_in[9];
  const float* b1   = (const float*)d_in[10];
  const float* W2   = (const float*)d_in[11];
  const float* b2   = (const float*)d_in[12];
  float* out = (float*)d_out;

  // workspace layout (bf16 / ushort elements)
  unsigned short* W1b  = (unsigned short*)d_ws;            // 128*256
  unsigned short* W2b  = W1b + C_P * E_TOT;                // 128*128
  unsigned short* pib  = W2b + C_P * C_P;                  // 512*64
  unsigned short* relT = pib + LSEQ * FEAT_DIM;            // 1024*64

  const int nconv = C_P * E_TOT + C_P * C_P;               // 49152
  convw_kernel<<<(nconv + 255) / 256, 256, 0, stream>>>(W1, W2, W1b, W2b);
  pi_kernel<<<LSEQ, 64, 0, stream>>>(s, W_sp, b_sp, pib);
  relpos_kernel<<<1024, 64, 0, stream>>>(W_rp, b_rp, relT);

  dim3 grid(LSEQ / 128, LSEQ);
  edge_main_kernel<<<grid, 256, 0, stream>>>(t, sc_t, pre, b1, b2,
                                             W1b, W2b, pib, relT, out);
}
